// EnhancedGNNTransformerEncoder_10385230922202
// MI455X (gfx1250) — matrix operand order repaired and checked by reading, not compiled, not run on hardware
//
#include <hip/hip_runtime.h>
#include <hip/hip_bf16.h>
#include <math.h>

// ---------------------------------------------------------------------------
// Types for CDNA5 WMMA (wave32): A/B fragments = 16 bf16 (8 VGPRs), C/D = 8 f32
// ---------------------------------------------------------------------------
typedef __attribute__((ext_vector_type(16))) __bf16   v16bf;
typedef __attribute__((ext_vector_type(8)))  float    v8f;
typedef __attribute__((ext_vector_type(4)))  unsigned v4u;
typedef __attribute__((ext_vector_type(2)))  unsigned v2u;

union Frag {
    v4u   u4[2];
    v16bf v;
};

__device__ __forceinline__ unsigned short f2bf(float f) {
    unsigned u = __float_as_uint(f);
    unsigned r = (u + 0x7FFFu + ((u >> 16) & 1u)) >> 16;   // round-to-nearest-even
    return (unsigned short)r;
}
__device__ __forceinline__ unsigned pack2bf(float lo, float hi) {
    return (unsigned)f2bf(lo) | ((unsigned)f2bf(hi) << 16);
}

// ---------------------------------------------------------------------------
// WMMA GEMM: Y[nrows, NOUT] = X[nrows, KDIM] @ W[KDIM, NOUT] + bias
// Block: 128 threads = 4 waves. Each wave owns MT=2 row tiles (32 rows), so a
// block covers 128 rows x NOUT cols. Each B (weight) fragment register load
// feeds MT WMMAs -> hot-loop DS traffic per WMMA drops ~2x vs MT=1.
// LDS: bf16 tiles, padded row stride 136 halves (272B, 16B multiple).
// ---------------------------------------------------------------------------
template <int KDIM, int NOUT, int MT>
__global__ __launch_bounds__(128)
void gemm_bias_kernel(const float* __restrict__ X, const float* __restrict__ W,
                      const float* __restrict__ bias, float* __restrict__ Y,
                      int nrows) {
    constexpr int NT   = NOUT / 16;     // column tiles per wave
    constexpr int ROWS = 64 * MT;       // rows per block
    __shared__ __align__(16) unsigned short Xs[ROWS * 136];
    __shared__ __align__(16) unsigned short Ws[NOUT * 136];

    const int tid  = threadIdx.x;
    const int lane = tid & 31;
    const int wv   = tid >> 5;          // 0..3
    const int blockRow = blockIdx.x * ROWS;

    v8f acc[MT][NT];
#pragma unroll
    for (int mt = 0; mt < MT; ++mt)
#pragma unroll
        for (int i = 0; i < NT; ++i)
            acc[mt][i] = (v8f){0.f,0.f,0.f,0.f,0.f,0.f,0.f,0.f};

    for (int k0 = 0; k0 < KDIM; k0 += 128) {
        // ---- stage X tile: float4 global loads, packed dword LDS stores ----
        // ROWS x 128 cols = ROWS*32 float4; 128 threads
#pragma unroll
        for (int j = 0; j < ROWS / 4; ++j) {
            int f4 = tid + j * 128;
            int r  = f4 >> 5;                // row
            int c4 = f4 & 31;                // float4 index within row
            int gr = blockRow + r;
            float4 val = make_float4(0.f, 0.f, 0.f, 0.f);
            if (gr < nrows)
                val = *(const float4*)&X[(size_t)gr * KDIM + k0 + c4 * 4];
            v2u p = (v2u){pack2bf(val.x, val.y), pack2bf(val.z, val.w)};
            *(v2u*)&Xs[r * 136 + c4 * 4] = p;
        }
        // ---- stage W^T tile: float4 global reads, transposed b16 stores ----
        for (int f4 = tid; f4 < 128 * NOUT / 4; f4 += 128) {
            int k  = f4 / (NOUT / 4);
            int n4 = f4 % (NOUT / 4);
            float4 w = *(const float4*)&W[(size_t)(k0 + k) * NOUT + n4 * 4];
            Ws[(n4 * 4 + 0) * 136 + k] = f2bf(w.x);
            Ws[(n4 * 4 + 1) * 136 + k] = f2bf(w.y);
            Ws[(n4 * 4 + 2) * 136 + k] = f2bf(w.z);
            Ws[(n4 * 4 + 3) * 136 + k] = f2bf(w.w);
        }
        __syncthreads();

        const int m  = lane & 15;
        const int kh = lane >> 4;       // half-wave selector
#pragma unroll
        for (int kk = 0; kk < 128; kk += 32) {
            Frag a[MT];
#pragma unroll
            for (int mt = 0; mt < MT; ++mt) {
                const unsigned short* ap =
                    &Xs[(wv * 16 * MT + mt * 16 + m) * 136 + kk + kh * 8];
                a[mt].u4[0] = *(const v4u*)(ap);        // K = kk + kh*8 .. +7
                a[mt].u4[1] = *(const v4u*)(ap + 16);   // K = kk + 16 + kh*8 .. +7
            }
#pragma unroll
            for (int ct = 0; ct < NT; ++ct) {
                Frag b;
                const unsigned short* wp = &Ws[(ct * 16 + m) * 136 + kk + kh * 8];
                b.u4[0] = *(const v4u*)(wp);        // K = kk + kh*8 .. +7
                b.u4[1] = *(const v4u*)(wp + 16);   // K = kk + 16 + kh*8 .. +7
#pragma unroll
                for (int mt = 0; mt < MT; ++mt)     // reuse B frag MT times
                    acc[mt][ct] = __builtin_amdgcn_wmma_f32_16x16x32_bf16(
                        false, a[mt].v, false, b.v, (short)0, acc[mt][ct],
                        false, false);
            }
        }
        __syncthreads();
    }

    // store: lane L -> col n = L&15, rows m = (L>>4)*8 + r
    const int n  = lane & 15;
    const int m0 = (lane >> 4) * 8;
#pragma unroll
    for (int mt = 0; mt < MT; ++mt)
#pragma unroll
        for (int ct = 0; ct < NT; ++ct) {
            float bv = bias[ct * 16 + n];
#pragma unroll
            for (int r = 0; r < 8; ++r) {
                int gr = blockRow + wv * 16 * MT + mt * 16 + m0 + r;
                if (gr < nrows)
                    Y[(size_t)gr * NOUT + ct * 16 + n] = acc[mt][ct][r] + bv;
            }
        }
}

// ---------------------------------------------------------------------------
// Small per-node kernels
// ---------------------------------------------------------------------------
__global__ void time_proj_kernel(const float* __restrict__ x,
                                 const float* __restrict__ Wt,
                                 const float* __restrict__ bt,
                                 float* __restrict__ out) {
    int nrow = blockIdx.x, d = threadIdx.x;   // 128 threads
    float x0 = x[nrow * 8 + 5], x1 = x[nrow * 8 + 6], x2 = x[nrow * 8 + 7];
    out[(size_t)nrow * 128 + d] =
        x0 * Wt[d] + x1 * Wt[128 + d] + x2 * Wt[256 + d] + bt[d];
}

__global__ void row_ln_kernel(const float* __restrict__ x,
                              const float* __restrict__ g,
                              const float* __restrict__ b,
                              float* __restrict__ y, int do_relu) {
    int nrow = blockIdx.x, t = threadIdx.x, lane = t & 31, wv = t >> 5;
    __shared__ float red[4];
    float v = x[(size_t)nrow * 128 + t];
    if (do_relu) v = fmaxf(v, 0.f);
    float sum = v;
#pragma unroll
    for (int o = 16; o > 0; o >>= 1) sum += __shfl_down(sum, o, 32);
    if (lane == 0) red[wv] = sum;
    __syncthreads();
    float mu = (red[0] + red[1] + red[2] + red[3]) * (1.f / 128.f);
    __syncthreads();
    float d = v - mu;
    float sq = d * d;
#pragma unroll
    for (int o = 16; o > 0; o >>= 1) sq += __shfl_down(sq, o, 32);
    if (lane == 0) red[wv] = sq;
    __syncthreads();
    float var = (red[0] + red[1] + red[2] + red[3]) * (1.f / 128.f);
    y[(size_t)nrow * 128 + t] = g[t] * d * rsqrtf(var + 1e-5f) + b[t];
}

__global__ void layer0_linear_kernel(const float* __restrict__ x,
    const float* __restrict__ Wq, const float* __restrict__ bq,
    const float* __restrict__ Wk, const float* __restrict__ bk,
    const float* __restrict__ Wv, const float* __restrict__ bv,
    const float* __restrict__ Wsk, const float* __restrict__ bsk,
    const float* __restrict__ Wp, const float* __restrict__ bp,
    float* __restrict__ q, float* __restrict__ k, float* __restrict__ v,
    float* __restrict__ sk, float* __restrict__ res) {
    int nrow = blockIdx.x, d = threadIdx.x;   // 128 threads
    __shared__ float xv[8];
    if (d < 8) xv[d] = x[nrow * 8 + d];
    __syncthreads();
    float aq = bq[d], ak = bk[d], av = bv[d], as = bsk[d], ar = bp[d];
#pragma unroll
    for (int i = 0; i < 8; ++i) {
        float xi = xv[i];
        aq += xi * Wq[i * 128 + d];
        ak += xi * Wk[i * 128 + d];
        av += xi * Wv[i * 128 + d];
        as += xi * Wsk[i * 128 + d];
        ar += xi * Wp[i * 128 + d];
    }
    size_t o = (size_t)nrow * 128 + d;
    q[o] = aq; k[o] = ak; v[o] = av; sk[o] = as; res[o] = ar;
}

// ---------------------------------------------------------------------------
// Edge kernels: scatter-softmax attention (one wave per edge, float4 loads)
// ---------------------------------------------------------------------------
__device__ __forceinline__ unsigned f2key(float f) {
    unsigned bits = __float_as_uint(f);
    return (bits & 0x80000000u) ? ~bits : (bits | 0x80000000u);
}
__device__ __forceinline__ float key2f(unsigned key) {
    unsigned bits = (key & 0x80000000u) ? (key & 0x7FFFFFFFu) : ~key;
    return __uint_as_float(bits);
}

__global__ void edge_alpha_max_kernel(const long long* __restrict__ ei,
                                      const float* __restrict__ q,
                                      const float* __restrict__ k,
                                      float* __restrict__ alpha,
                                      unsigned* __restrict__ mkey, int nedges) {
    int e = blockIdx.x * 8 + (threadIdx.x >> 5);
    int lane = threadIdx.x & 31;
    if (e >= nedges) return;
    int src = (int)ei[e];
    int dst = (int)ei[nedges + e];
    float4 qv = ((const float4*)(q + (size_t)dst * 128))[lane];
    float4 kv = ((const float4*)(k + (size_t)src * 128))[lane];
    float sum = qv.x * kv.x + qv.y * kv.y + qv.z * kv.z + qv.w * kv.w;
#pragma unroll
    for (int o = 16; o > 0; o >>= 1) sum += __shfl_down(sum, o, 32);
    if (lane == 0) {
        float a = sum * 0.08838834764831845f;   // 1/sqrt(128)
        alpha[e] = a;
        atomicMax(&mkey[dst], f2key(a));
    }
}

__global__ void edge_accum_kernel(const long long* __restrict__ ei,
                                  const float* __restrict__ v,
                                  const float* __restrict__ alpha,
                                  const unsigned* __restrict__ mkey,
                                  float* __restrict__ s,
                                  float* __restrict__ agg, int nedges) {
    int e = blockIdx.x * 8 + (threadIdx.x >> 5);
    int lane = threadIdx.x & 31;
    if (e >= nedges) return;
    int src = (int)ei[e];
    int dst = (int)ei[nedges + e];
    float m  = key2f(mkey[dst]);
    float ex = __expf(alpha[e] - m);
    if (lane == 0) atomicAdd(&s[dst], ex);
    float4 vv = ((const float4*)(v + (size_t)src * 128))[lane];
    float* ad = agg + (size_t)dst * 128 + lane * 4;
    atomicAdd(ad + 0, vv.x * ex);
    atomicAdd(ad + 1, vv.y * ex);
    atomicAdd(ad + 2, vv.z * ex);
    atomicAdd(ad + 3, vv.w * ex);
}

__global__ void init_accum_kernel(float* __restrict__ agg, float* __restrict__ s,
                                  unsigned* __restrict__ mkey, int nnodes) {
    size_t i = (size_t)blockIdx.x * 256 + threadIdx.x;
    if (i < (size_t)nnodes * 128) agg[i] = 0.f;
    if (i < (size_t)nnodes) { s[i] = 0.f; mkey[i] = 0u; }
}

__global__ void node_combine_kernel(const float* __restrict__ agg,
                                    const float* __restrict__ s,
                                    const float* __restrict__ sk,
                                    const float* __restrict__ prev,
                                    float* __restrict__ out, int nnodes) {
    size_t i = (size_t)blockIdx.x * 256 + threadIdx.x;
    if (i >= (size_t)nnodes * 128) return;
    int node = (int)(i >> 7);
    float sv = s[node];
    if (sv == 0.f) sv = 1.f;
    float t = agg[i] / sv + sk[i];
    out[i] = fmaxf(t, 0.f) + prev[i];
}

__global__ void concat_kernel(const float* __restrict__ a,
                              const float* __restrict__ b,
                              float* __restrict__ out, int nnodes) {
    size_t i = (size_t)blockIdx.x * 256 + threadIdx.x;
    if (i >= (size_t)nnodes * 256) return;
    int node = (int)(i >> 8);
    int c = (int)(i & 255);
    out[i] = (c < 128) ? a[(size_t)node * 128 + c]
                       : b[(size_t)node * 128 + (c - 128)];
}

// ---------------------------------------------------------------------------
// Host driver
// ---------------------------------------------------------------------------
extern "C" void kernel_launch(void* const* d_in, const int* in_sizes, int n_in,
                              void* d_out, int out_size, void* d_ws, size_t ws_size,
                              hipStream_t stream) {
    const int N = in_sizes[2];          // batch vector length == #nodes
    const int E = in_sizes[1] / 2;

    // inputs (setup_inputs order)
    const float*     x    = (const float*)d_in[0];
    const long long* ei   = (const long long*)d_in[1];
    const float* Wt  = (const float*)d_in[3];  const float* bt  = (const float*)d_in[4];
    const float* ltg = (const float*)d_in[5];  const float* ltb = (const float*)d_in[6];
    // spatial encoder (d_in[7..10]) is computed-but-unused in reference: skip.
    const float* Wq0 = (const float*)d_in[11]; const float* bq0 = (const float*)d_in[12];
    const float* Wk0 = (const float*)d_in[13]; const float* bk0 = (const float*)d_in[14];
    const float* Wv0 = (const float*)d_in[15]; const float* bv0 = (const float*)d_in[16];
    const float* Ws0 = (const float*)d_in[17]; const float* bs0 = (const float*)d_in[18];
    const float* Wp0 = (const float*)d_in[19]; const float* bp0 = (const float*)d_in[20];
    const float* Wq  = (const float*)d_in[21]; const float* bq  = (const float*)d_in[22];
    const float* Wk  = (const float*)d_in[23]; const float* bk  = (const float*)d_in[24];
    const float* Wv  = (const float*)d_in[25]; const float* bv  = (const float*)d_in[26];
    const float* Wsk = (const float*)d_in[27]; const float* bsk = (const float*)d_in[28];
    const float* Wf  = (const float*)d_in[29]; const float* bf  = (const float*)d_in[30];
    const float* lfg = (const float*)d_in[31]; const float* lfb = (const float*)d_in[32];
    const float* Wo  = (const float*)d_in[33]; const float* bo  = (const float*)d_in[34];
    float* out = (float*)d_out;

    // workspace carve-up
    float* f = (float*)d_ws;
    const size_t ND = (size_t)N * 128;
    float* qb   = f; f += ND;
    float* kb   = f; f += ND;
    float* vb   = f; f += ND;
    float* skb  = f; f += ND;
    float* agg  = f; f += ND;
    float* curA = f; f += ND;
    float* curB = f; f += ND;
    float* tenc = f; f += ND;
    float* comb = f; f += 2 * ND;
    float* alph = f; f += (size_t)E;
    float* sbuf = f; f += (size_t)N;
    unsigned* mkey = (unsigned*)f;

    const dim3 blk128(128), blk256(256);
    const unsigned gNode   = (unsigned)N;
    const unsigned gElem   = (unsigned)(((size_t)N * 128 + 255) / 256);
    const unsigned gConcat = (unsigned)(((size_t)N * 256 + 255) / 256);
    const unsigned gEdge   = (unsigned)((E + 7) / 8);
    const unsigned gGemm   = (unsigned)((N + 127) / 128);   // 128 rows per block

    // ---- time encoder: tenc = LN(relu(x[:,5:8] @ Wt + bt)) ----
    time_proj_kernel<<<gNode, blk128, 0, stream>>>(x, Wt, bt, curA);
    row_ln_kernel<<<gNode, blk128, 0, stream>>>(curA, ltg, ltb, tenc, 1);

    // ---- layer 0 (K=8, VALU path) ----
    layer0_linear_kernel<<<gNode, blk128, 0, stream>>>(
        x, Wq0, bq0, Wk0, bk0, Wv0, bv0, Ws0, bs0, Wp0, bp0,
        qb, kb, vb, skb, curB /* residual */);
    init_accum_kernel<<<gElem, blk256, 0, stream>>>(agg, sbuf, mkey, N);
    edge_alpha_max_kernel<<<gEdge, blk256, 0, stream>>>(ei, qb, kb, alph, mkey, E);
    edge_accum_kernel<<<gEdge, blk256, 0, stream>>>(ei, vb, alph, mkey, sbuf, agg, E);
    node_combine_kernel<<<gElem, blk256, 0, stream>>>(agg, sbuf, skb, curB, curA, N);

    // ---- layers 1..7 (WMMA GEMMs, MT=2) ----
    float* cur = curA;
    float* nxt = curB;
    for (int L = 0; L < 7; ++L) {
        const size_t wo  = (size_t)L * 128 * 128;
        const size_t bo_ = (size_t)L * 128;
        gemm_bias_kernel<128, 128, 2><<<gGemm, blk128, 0, stream>>>(cur, Wq  + wo, bq  + bo_, qb,  N);
        gemm_bias_kernel<128, 128, 2><<<gGemm, blk128, 0, stream>>>(cur, Wk  + wo, bk  + bo_, kb,  N);
        gemm_bias_kernel<128, 128, 2><<<gGemm, blk128, 0, stream>>>(cur, Wv  + wo, bv  + bo_, vb,  N);
        gemm_bias_kernel<128, 128, 2><<<gGemm, blk128, 0, stream>>>(cur, Wsk + wo, bsk + bo_, skb, N);
        init_accum_kernel<<<gElem, blk256, 0, stream>>>(agg, sbuf, mkey, N);
        edge_alpha_max_kernel<<<gEdge, blk256, 0, stream>>>(ei, qb, kb, alph, mkey, E);
        edge_accum_kernel<<<gEdge, blk256, 0, stream>>>(ei, vb, alph, mkey, sbuf, agg, E);
        node_combine_kernel<<<gElem, blk256, 0, stream>>>(agg, sbuf, skb, cur, nxt, N);
        float* t = cur; cur = nxt; nxt = t;
    }

    // ---- fusion + output ----
    concat_kernel<<<gConcat, blk256, 0, stream>>>(cur, tenc, comb, N);
    gemm_bias_kernel<256, 128, 2><<<gGemm, blk128, 0, stream>>>(comb, Wf, bf, qb, N);
    row_ln_kernel<<<gNode, blk128, 0, stream>>>(qb, lfg, lfb, kb, 1);
    gemm_bias_kernel<128, 64, 2><<<gGemm, blk128, 0, stream>>>(kb, Wo, bo, out, N);
}
